// MoTAttention_58832462020712
// MI455X (gfx1250) — matrix operand order repaired and checked by reading, not compiled, not run on hardware
//
#include <hip/hip_runtime.h>
#include <stdint.h>

// ---------- CDNA5 WMMA plumbing ----------
typedef __attribute__((ext_vector_type(16))) __bf16 v16bf;
typedef __attribute__((ext_vector_type(8)))  float  v8f;

union FragU { uint32_t u[8]; v16bf v; };

__device__ __forceinline__ uint32_t f2bf_u16(float f) {
  union { float f; uint32_t u; } c; c.f = f;
  uint32_t u = c.u;
  return (u + 0x7FFFu + ((u >> 16) & 1u)) >> 16;   // RNE
}
__device__ __forceinline__ uint32_t pkbf(float a, float b) {
#if __has_builtin(__builtin_amdgcn_cvt_pk_bf16_f32)
  typedef __attribute__((ext_vector_type(2))) __bf16 v2bf;
  union { v2bf v; uint32_t u; } c;
  c.v = __builtin_amdgcn_cvt_pk_bf16_f32(a, b);    // a -> lo, b -> hi
  return c.u;
#else
  return f2bf_u16(a) | (f2bf_u16(b) << 16);
#endif
}
// A-matrix (16x32 bf16) K index for packed pair v (v=0..7), per ISA 7.12.2
__device__ __forceinline__ int kA(int v, int lane) {
  return ((v & 3) << 1) + ((v >> 2) << 4) + (((lane >> 4) & 1) << 3);
}
// B-matrix (32x16 bf16) K index for packed pair v, same K order as kA
__device__ __forceinline__ int kB(int v, int lane) {
  return ((v & 3) << 1) + ((v >> 2) << 4) + (((lane >> 4) & 1) << 3);
}
__device__ __forceinline__ v8f wmma_bf16(const FragU& a, const FragU& b, v8f c) {
  return __builtin_amdgcn_wmma_f32_16x16x32_bf16(false, a.v, false, b.v,
                                                 (short)0, c, false, false);
}

// ---------- CDNA5 async global->LDS (ASYNCcnt path), non-temporal ----------
__device__ __forceinline__ void async_g2l_b128(void* lds, const void* g) {
  uint32_t l = (uint32_t)(uintptr_t)lds;   // low 32 bits of generic shared ptr = LDS offset
  uint64_t a = (uint64_t)(uintptr_t)g;
  asm volatile("global_load_async_to_lds_b128 %0, %1, off th:TH_LOAD_NT"
               :: "v"(l), "v"(a) : "memory");
}
__device__ __forceinline__ void wait_async0() {
#if __has_builtin(__builtin_amdgcn_s_wait_asynccnt)
  __builtin_amdgcn_s_wait_asynccnt(0);
#else
  asm volatile("s_wait_asynccnt 0" ::: "memory");
#endif
}

// ---------- problem constants ----------
#define MOT_B  2
#define MOT_L  1024
#define MOT_D  768
#define MOT_H  12
#define MOT_HP 16                       // padded head count for LDS score buffer
#define MOT_DH 64
#define RC     32                       // r-chunk
#define MOT_SCALE 0.07216878364870323f  // 1/sqrt(3*64)

// =====================================================================
// GEMM: C[M,N] = A[M,K] @ (TRANSB ? W[N,K]^T : W[K,N]) (+bias)
// One 16x16 C tile per wave, bf16 WMMA 16x16x32, f32 accumulate.
// Optionally also emits a bf16 (u16) copy of C for downstream WMMA use.
// =====================================================================
template <bool TRANSB>
__global__ __launch_bounds__(256) void mot_gemm(
    const float* __restrict__ A, const float* __restrict__ W,
    const float* __restrict__ bias, float* __restrict__ C,
    uint16_t* __restrict__ Cb, int M, int N, int K)
{
  const int lane = threadIdx.x & 31;
  const int wv   = threadIdx.x >> 5;
  const int hi   = (lane >> 4) & 1;
  const int lr   = lane & 15;
  const int tilesN = N >> 4;
  const int tiles  = (M >> 4) * tilesN;
  const int nwaves = (gridDim.x * blockDim.x) >> 5;

  for (int t = blockIdx.x * 8 + wv; t < tiles; t += nwaves) {
    const int m0 = (t / tilesN) << 4;
    const int n0 = (t % tilesN) << 4;
    const int row = m0 + lr, col = n0 + lr;
    const float* ap = A + (long)row * K;
    const float* wp = TRANSB ? (W + (long)col * K) : (W + col);
    v8f acc = {};
    for (int kk = 0; kk < K; kk += 32) {
      FragU a, b;
#pragma unroll
      for (int v = 0; v < 8; ++v) {
        const int ka = kk + kA(v, lane);
        const int kb = kk + kB(v, lane);
        a.u[v] = pkbf(ap[ka], ap[ka + 1]);
        if (TRANSB)
          b.u[v] = pkbf(wp[kb], wp[kb + 1]);
        else
          b.u[v] = pkbf(wp[(long)kb * N], wp[(long)(kb + 1) * N]);
      }
      acc = wmma_bf16(a, b, acc);
    }
#pragma unroll
    for (int v = 0; v < 8; ++v) {
      const int mr = m0 + v + (hi << 3);
      float o = acc[v];
      if (bias) o += bias[col];
      C[(long)mr * N + col] = o;
      if (Cb) Cb[(long)mr * N + col] = (uint16_t)f2bf_u16(o);
    }
  }
}

// =====================================================================
// Fused relative attention. One workgroup per (b, 16-row l-tile); all 12
// heads share each rel chunk (rel streamed from HBM exactly once, via
// async DMA, software-pipelined against the WMMA work). Online softmax.
// Q/K/V/RQ/RK arrive pre-packed bf16.
// =====================================================================
__global__ __launch_bounds__(256) void mot_attn(
    const uint16_t* __restrict__ Qb,  const uint16_t* __restrict__ Kb,
    const uint16_t* __restrict__ Vb,  const uint16_t* __restrict__ RQb,
    const uint16_t* __restrict__ RKb, const float* __restrict__ relg,
    const float* __restrict__ maskg,  float* __restrict__ Og)
{
  extern __shared__ float smem[];
  float*    s_rel  = smem;                          // f32 stage [16][32][64] = 32768 f
  uint32_t* s_relb = (uint32_t*)(s_rel + 16 * RC * MOT_DH);     // bf16 pairs, 16384 u32
  float*    s_O    = (float*)(s_relb + 16 * RC * (MOT_DH / 2)); // [12][16][64] = 12288 f
  float*    s_S    = s_O + MOT_H * 16 * MOT_DH;     // [16][16][32] padded heads = 8192 f
  uint32_t* s_P    = (uint32_t*)(s_S + MOT_HP * 16 * RC);       // 3072 u32
  float*    s_m    = (float*)(s_P + MOT_H * 16 * (RC / 2));     // 192
  float*    s_sum  = s_m + 192;
  float*    s_corr = s_sum + 192;

  const int tid  = threadIdx.x;
  const int lane = tid & 31, wv = tid >> 5;
  const int hi   = (lane >> 4) & 1, lr = lane & 15;
  const int b    = blockIdx.x >> 6;
  const int l0   = (blockIdx.x & 63) << 4;

  for (int i = tid; i < MOT_H * 16 * MOT_DH; i += 256) s_O[i] = 0.f;
  if (tid < 192) { s_m[tid] = -1e30f; s_sum[tid] = 0.f; }
  __syncthreads();

  // issue async DMA of a rel chunk: rel[b, l0:+16, r0:+32, :] -> s_rel (128 KB)
  auto issue_rel = [&](int r0) {
    for (int it = 0; it < 32; ++it) {
      const int id  = tid + (it << 8);      // 8192 x 16B segments
      const int li  = id >> 9;
      const int rem = id & 511;
      const int rj  = rem >> 4;
      const int q4  = (rem & 15) << 2;
      const float* gp = relg +
          (((long)(b * MOT_L + l0 + li) * MOT_L + (r0 + rj)) * MOT_DH + q4);
      async_g2l_b128(s_rel + ((li * RC + rj) * MOT_DH + q4), gp);
    }
  };

  issue_rel(0);

  for (int ri = 0; ri < MOT_L / RC; ++ri) {
    const int r0 = ri * RC;

    wait_async0();
    __syncthreads();

    // ---- convert staged rel f32 -> packed bf16 (each element exactly once) ----
    for (int i = tid; i < 16 * RC * (MOT_DH / 2); i += 256)
      s_relb[i] = pkbf(s_rel[2 * i], s_rel[2 * i + 1]);
    __syncthreads();

    // f32 stage is now dead: overlap next chunk's DMA with this chunk's WMMAs
    if (ri + 1 < MOT_L / RC) issue_rel(r0 + RC);

    // ---- phase 1: 24 q.k^T jobs; exclusive writers -> plain stores, no zeroing ----
    for (int j = wv; j < 24; j += 8) {
      const int h = j >> 1, nh = j & 1;
      const uint32_t* qp = (const uint32_t*)Qb +
          (((long)(b * MOT_L + l0 + lr) * MOT_D + h * MOT_DH) >> 1);
      const uint32_t* kp = (const uint32_t*)Kb +
          (((long)(b * MOT_L + r0 + nh * 16 + lr) * MOT_D + h * MOT_DH) >> 1);
      v8f acc = {};
      for (int kk = 0; kk < MOT_DH; kk += 32) {
        FragU a, bb;
#pragma unroll
        for (int v = 0; v < 8; ++v) {
          a.u[v]  = qp[(kk + kA(v, lane)) >> 1];
          bb.u[v] = kp[(kk + kB(v, lane)) >> 1];
        }
        acc = wmma_bf16(a, bb, acc);
      }
#pragma unroll
      for (int v = 0; v < 8; ++v) {
        const int ml = v + (hi << 3);
        s_S[(h * 16 + ml) * RC + nh * 16 + lr] = acc[v];
      }
    }
    __syncthreads();

    // ---- phase 2: 64 rel-term jobs, unconditional ds_add into padded S ----
    const int hr = (lr < MOT_H) ? lr : 0;
    const uint32_t hmask = (lr < MOT_H) ? 0xFFFFFFFFu : 0u;
    for (int j = wv; j < 64; j += 8) {
      v8f acc = {};
      if (j < 32) {                    // rq_l . rel_l^T : C[h, r] for fixed l
        const int li = j >> 1, nh = j & 1;
        const uint32_t* rqp = (const uint32_t*)RQb +
            (((long)(b * MOT_L + l0 + li) * MOT_D) >> 1);
        const uint32_t* rp = s_relb + (li * RC + nh * 16 + lr) * (MOT_DH / 2);
        for (int kk = 0; kk < MOT_DH; kk += 32) {
          FragU a, bb;
#pragma unroll
          for (int v = 0; v < 8; ++v) {
            a.u[v]  = rqp[hr * (MOT_DH / 2) + ((kk + kA(v, lane)) >> 1)] & hmask;
            bb.u[v] = rp[(kk + kB(v, lane)) >> 1];
          }
          acc = wmma_bf16(a, bb, acc);
        }
#pragma unroll
        for (int v = 0; v < 8; ++v) {
          const int mh = v + (hi << 3);  // rows >= 12 land in padded scratch
          atomicAdd(&s_S[(mh * 16 + li) * RC + nh * 16 + lr], acc[v]);
        }
      } else {                         // rk_r . rel_r^T : C[h, l] for fixed r
        const int r = j - 32;
        const uint32_t* rkp = (const uint32_t*)RKb +
            (((long)(b * MOT_L + r0 + r) * MOT_D) >> 1);
        const uint32_t* rp = s_relb + (lr * RC + r) * (MOT_DH / 2); // B[k][l=lr]
        for (int kk = 0; kk < MOT_DH; kk += 32) {
          FragU a, bb;
#pragma unroll
          for (int v = 0; v < 8; ++v) {
            a.u[v]  = rkp[hr * (MOT_DH / 2) + ((kk + kA(v, lane)) >> 1)] & hmask;
            bb.u[v] = rp[(kk + kB(v, lane)) >> 1];
          }
          acc = wmma_bf16(a, bb, acc);
        }
#pragma unroll
        for (int v = 0; v < 8; ++v) {
          const int mh = v + (hi << 3);  // rows >= 12 land in padded scratch
          atomicAdd(&s_S[(mh * 16 + lr) * RC + r], acc[v]);
        }
      }
    }
    __syncthreads();

    // ---- online softmax update, one thread per (h, l) ----
    if (tid < 192) {
      const int hl = tid;
      const float m_old = s_m[hl];
      float vals[RC];
      float mc = -1e30f;
#pragma unroll
      for (int r = 0; r < RC; ++r) {
        const float x = s_S[hl * RC + r] * MOT_SCALE + maskg[b * MOT_L + r0 + r];
        vals[r] = x;
        mc = fmaxf(mc, x);
      }
      const float m_new = fmaxf(m_old, mc);
      const float corr  = __expf(m_old - m_new);
      float sum = s_sum[hl] * corr;
#pragma unroll
      for (int r = 0; r < RC; r += 2) {
        const float p0 = __expf(vals[r] - m_new);
        const float p1 = __expf(vals[r + 1] - m_new);
        sum += p0 + p1;
        s_P[hl * (RC / 2) + (r >> 1)] = pkbf(p0, p1);
      }
      s_m[hl] = m_new; s_sum[hl] = sum; s_corr[hl] = corr;
    }
    __syncthreads();

    // ---- 48 PV WMMA jobs: O[h, l, dtile] = O*corr + P[h] @ V[h] ----
    for (int j = wv; j < 48; j += 8) {
      const int h = j >> 2, dt = j & 3;
      FragU a, bb;
      v8f acc = {};
#pragma unroll
      for (int v = 0; v < 8; ++v) {
        const int ka = kA(v, lane);          // even -> packed pair index
        const int kb = kB(v, lane);
        a.u[v] = s_P[(h * 16 + lr) * (RC / 2) + (ka >> 1)];
        const uint16_t* vp = Vb + (long)(b * MOT_L + r0 + kb) * MOT_D
                                + h * MOT_DH + dt * 16 + lr;
        bb.u[v] = (uint32_t)vp[0] | ((uint32_t)vp[MOT_D] << 16);
      }
      acc = wmma_bf16(a, bb, acc);
#pragma unroll
      for (int v = 0; v < 8; ++v) {
        const int ml = v + (hi << 3);
        const int o  = (h * 16 + ml) * MOT_DH + dt * 16 + lr;
        s_O[o] = s_O[o] * s_corr[h * 16 + ml] + acc[v];
      }
    }
    __syncthreads();
  }

  // ---- normalize and emit attended[b, l, h, d] (f32) ----
  if (tid < 192) {
    const int h = tid >> 4, li = tid & 15;
    const float inv = 1.0f / s_sum[tid];
    float* op = Og + (long)(b * MOT_L + l0 + li) * MOT_D + h * MOT_DH;
    for (int q = 0; q < MOT_DH; ++q) op[q] = s_O[tid * MOT_DH + q] * inv;
  }
}

// =====================================================================
extern "C" void kernel_launch(void* const* d_in, const int* in_sizes, int n_in,
                              void* d_out, int out_size, void* d_ws, size_t ws_size,
                              hipStream_t stream) {
  (void)in_sizes; (void)n_in; (void)out_size; (void)ws_size;
  const float* hs   = (const float*)d_in[0];
  const float* mask = (const float*)d_in[1];
  const float* rel  = (const float*)d_in[2];
  const float* Wq   = (const float*)d_in[3];
  const float* Wk   = (const float*)d_in[4];
  const float* Wv   = (const float*)d_in[5];
  const float* bv   = (const float*)d_in[6];
  const float* Wo   = (const float*)d_in[7];
  const float* bo   = (const float*)d_in[8];
  float* out = (float*)d_out;

  const int M = MOT_B * MOT_L, N = MOT_D, K = MOT_D;
  const size_t sz = (size_t)M * N;
  // f32 workspace
  float* Q  = (float*)d_ws;
  float* Kt = Q  + sz;
  float* V  = Kt + sz;
  float* RQ = V  + sz;
  float* RK = RQ + sz;
  float* AT = RK + sz;   // attended, [B,L,D]
  // packed-bf16 workspace
  uint16_t* Qb  = (uint16_t*)(AT + sz);
  uint16_t* Kb  = Qb  + sz;
  uint16_t* Vb  = Kb  + sz;
  uint16_t* RQb = Vb  + sz;
  uint16_t* RKb = RQb + sz;

  const dim3 gblk(256), ggrd((M / 16) * (N / 16) / 8);
  mot_gemm<true ><<<ggrd, gblk, 0, stream>>>(hs, Wq, nullptr, Q,  Qb,  M, N, K);
  mot_gemm<true ><<<ggrd, gblk, 0, stream>>>(hs, Wk, nullptr, Kt, Kb,  M, N, K);
  mot_gemm<true ><<<ggrd, gblk, 0, stream>>>(hs, Wv, bv,      V,  Vb,  M, N, K);
  mot_gemm<false><<<ggrd, gblk, 0, stream>>>(Q,  Wk, nullptr, RQ, RQb, M, N, K);
  mot_gemm<false><<<ggrd, gblk, 0, stream>>>(Kt, Wq, nullptr, RK, RKb, M, N, K);

  const size_t smem_bytes =
      (16 * RC * MOT_DH) * 4              // f32 rel stage
      + (16 * RC * (MOT_DH / 2)) * 4      // bf16 rel tile
      + (MOT_H * 16 * MOT_DH) * 4         // O accumulators
      + (MOT_HP * 16 * RC) * 4            // S scores (padded to 16 heads)
      + (MOT_H * 16 * (RC / 2)) * 4       // P packed probs
      + 3 * 192 * 4;                      // m / sum / corr   (~287 KB < 320 KB)
  mot_attn<<<dim3(MOT_B * (MOT_L / 16)), dim3(256), smem_bytes, stream>>>(
      Qb, Kb, Vb, RQb, RKb, rel, mask, AT);

  mot_gemm<true ><<<ggrd, gblk, 0, stream>>>(AT, Wo, bo, out, nullptr, M, N, K);
}
